// CRF_23210003267718
// MI455X (gfx1250) — compile-verified, matrix operand order checked by reading, not run on hardware
//
#include <hip/hip_runtime.h>

typedef _Float16 v16h __attribute__((ext_vector_type(16)));
typedef _Float16 v8h  __attribute__((ext_vector_type(8)));
typedef float    v8f  __attribute__((ext_vector_type(8)));

#define NUM_E   64
#define BATCH   512
#define SEQ     512
#define BOS_ID  1
#define EOS_ID  2
#define LOG2E   1.44269504088896340736f
#define LN2     0.69314718055994530942f
#define AH      72     // f16 row stride: 144B rows -> 16B aligned

__device__ __forceinline__ float fexp(float x) {   // e^x via v_exp_f32
    return __builtin_amdgcn_exp2f(x * LOG2E);
}
__device__ __forceinline__ float flog(float x) {   // ln(x) via v_log_f32
    return __builtin_amdgcn_logf(x) * LN2;
}

// K index inside a 16-bit WMMA operand fragment (16 halves/lane covers K 0..31)
__device__ __forceinline__ int kmap(int h, int hi) {
    return h + ((h >= 8) ? 8 : 0) + (hi ? 8 : 0);
}

// ---------------------------------------------------------------------------
// Forward algorithm — one wave32 per 16 batch rows, exp-space the whole way:
//   q      = D * exp(em)                (D = A_prev @ exp(T), WMMA f16->f32)
//   E      = exponent(rowmax(q))        (bit trick, exact)
//   A_next = q * 2^-E   (f16, in LDS)   off += E * ln2
// LDS ops from one wave are in-order (ISA: DScnt "in order"), so the
// store->load handoff between steps needs only a compiler barrier; the
// ds_load -> WMMA VGPR dependency gets its s_wait_dscnt from the compiler.
// ---------------------------------------------------------------------------
__launch_bounds__(32)
__global__ void crf_forward_scan(const float* __restrict__ em,
                                 const float* __restrict__ Tr,
                                 float* __restrict__ logz) {
    __shared__ __align__(16) _Float16 alphaH[16 * AH];  // normalized exp(alpha-off)
    __shared__ float offS[16];

    const int lane = threadIdx.x;       // wave32
    const int hi   = lane >> 4;
    const int lq   = lane & 15;
    const int b0   = blockIdx.x * 16;

    // ---- resident B fragments: exp(T), four 64x16 column tiles, f16 ----
    v16h Bf[2][4];
#pragma unroll
    for (int kh = 0; kh < 2; ++kh) {
#pragma unroll
        for (int j = 0; j < 4; ++j) {
            v16h v;
#pragma unroll
            for (int h = 0; h < 16; ++h) {
                int k = kmap(h, hi) + 32 * kh;
                int n = lq + 16 * j;
                v[h] = (_Float16)fexp(Tr[k * NUM_E + n]);
            }
            Bf[kh][j] = v;
        }
    }

    // Per-lane emission base; rows r via immediate offsets (r*131072 B < 2^23).
    const float* embase = em + ((size_t)(b0 + 8 * hi) * SEQ) * NUM_E + lq;

    // ---- t = 0 : alpha0 = T[BOS,:] + em[:,0,:]; store exp(alpha0 - max) ----
    float offv[8];
#pragma unroll
    for (int r = 0; r < 8; ++r) {
        int m = r + 8 * hi;
        float a[4], mx = -3.0e38f;
#pragma unroll
        for (int j = 0; j < 4; ++j) {
            a[j] = Tr[BOS_ID * NUM_E + lq + 16 * j]
                 + embase[r * (SEQ * NUM_E) + 16 * j];
            mx = fmaxf(mx, a[j]);
        }
#pragma unroll
        for (int s = 1; s < 16; s <<= 1)
            mx = fmaxf(mx, __shfl_xor(mx, s, 16));
#pragma unroll
        for (int j = 0; j < 4; ++j)
            alphaH[m * AH + lq + 16 * j] = (_Float16)fexp(a[j] - mx);
        offv[r] = mx;
    }

    const v8f zero8 = {};
    const float* ep = embase + NUM_E;   // step t emissions

    for (int t = 1; t < SEQ; ++t, ep += NUM_E) {
        // ---- early emission loads; exp(em) hides under the WMMA chain ----
        float expem[8][4];
#pragma unroll
        for (int r = 0; r < 8; ++r)
#pragma unroll
            for (int j = 0; j < 4; ++j)
                expem[r][j] = ep[r * (SEQ * NUM_E) + 16 * j];
#pragma unroll
        for (int r = 0; r < 8; ++r)
#pragma unroll
            for (int j = 0; j < 4; ++j)
                expem[r][j] = fexp(expem[r][j]);

        // stores (prev step) -> loads (this step): order only; LDS is in-order
        asm volatile("" ::: "memory");

        // ---- A fragment: 4 x ds_load_b128 of ready f16 halves ----
        const _Float16* rowp = alphaH + lq * AH;
        v8h p0 = *(const v8h*)(rowp + 8 * hi);        // k =  8hi..
        v8h p1 = *(const v8h*)(rowp + 16 + 8 * hi);   // k = 16+8hi..
        v8h p2 = *(const v8h*)(rowp + 32 + 8 * hi);
        v8h p3 = *(const v8h*)(rowp + 48 + 8 * hi);
        v16h Alo = __builtin_shufflevector(p0, p1, 0,1,2,3,4,5,6,7,
                                           8,9,10,11,12,13,14,15);
        v16h Ahi = __builtin_shufflevector(p2, p3, 0,1,2,3,4,5,6,7,
                                           8,9,10,11,12,13,14,15);

        // ---- D_j = A(16x64) x expT(64x16 tile j), chained K=32 chunks ----
        v8f D[4];
#pragma unroll
        for (int j = 0; j < 4; ++j) {
            v8f acc = __builtin_amdgcn_wmma_f32_16x16x32_f16(
                false, Alo, false, Bf[0][j], (short)0, zero8, false, false);
            D[j] = __builtin_amdgcn_wmma_f32_16x16x32_f16(
                false, Ahi, false, Bf[1][j], (short)0, acc, false, false);
        }

        // ---- q = D*exp(em); power-of-2 renormalize; store f16 ----
#pragma unroll
        for (int r = 0; r < 8; ++r) {
            int m = r + 8 * hi;
            float q0 = D[0][r] * expem[r][0];
            float q1 = D[1][r] * expem[r][1];
            float q2 = D[2][r] * expem[r][2];
            float q3 = D[3][r] * expem[r][3];
            float mx = fmaxf(fmaxf(q0, q1), fmaxf(q2, q3));
#pragma unroll
            for (int s = 1; s < 16; s <<= 1)
                mx = fmaxf(mx, __shfl_xor(mx, s, 16));
            // exponent-only normalization: exact, no rcp / no log
            unsigned qb  = __float_as_uint(mx) & 0x7F800000u;
            float scale  = __uint_as_float((254u << 23) - qb);   // 2^-E
            int   E      = (int)(qb >> 23) - 127;
            offv[r] += (float)E * LN2;
            alphaH[m * AH + lq + 16 * 0] = (_Float16)(q0 * scale);
            alphaH[m * AH + lq + 16 * 1] = (_Float16)(q1 * scale);
            alphaH[m * AH + lq + 16 * 2] = (_Float16)(q2 * scale);
            alphaH[m * AH + lq + 16 * 3] = (_Float16)(q3 * scale);
        }

        // prefetch a future emission line (rotates across the 8 rows)
        if (t + 8 < SEQ) {
            int pr = t & 7;
            __builtin_prefetch((const void*)(ep + 8 * NUM_E
                                             + pr * (SEQ * NUM_E) - lq + lq * 4),
                               0, 1);
        }
    }

    // ---- publish off, then log_z[m] = off + log( sum_k A[m,k]*exp(T[k,EOS]) )
    if (lane == 0) {
#pragma unroll
        for (int r = 0; r < 8; ++r) offS[r] = offv[r];
    }
    if (lane == 16) {
#pragma unroll
        for (int r = 0; r < 8; ++r) offS[8 + r] = offv[r];
    }
    asm volatile("s_wait_dscnt 0" ::: "memory");

    if (lane < 16) {
        float offm = offS[lane];
        float s = 0.0f;
#pragma unroll
        for (int k = 0; k < NUM_E; ++k)
            s += (float)alphaH[lane * AH + k] * fexp(Tr[k * NUM_E + EOS_ID]);
        logz[b0 + lane] = offm + flog(s);
    }
}

// ---------------------------------------------------------------------------
// Gold path score (gathers), one block per batch row.
// ---------------------------------------------------------------------------
__launch_bounds__(256)
__global__ void crf_gold_score(const float* __restrict__ em,
                               const float* __restrict__ Tr,
                               const int* __restrict__ ent,
                               const unsigned char* __restrict__ mask,
                               float* __restrict__ score) {
    __shared__ float ssum[256];
    __shared__ int   scnt[256];
    const int b = blockIdx.x, tid = threadIdx.x;

    float acc = 0.0f;
    int   cnt = 0;
    for (int t = tid; t < SEQ; t += 256) {
        int mt = mask[(size_t)b * SEQ + t] ? 1 : 0;
        cnt += mt;
        if (t >= 1) {
            int e  = ent[(size_t)b * SEQ + t];
            int pe = ent[(size_t)b * SEQ + t - 1];
            float term = em[((size_t)b * SEQ + t) * NUM_E + e] + Tr[pe * NUM_E + e];
            acc += mt ? term : 0.0f;
        }
    }
    ssum[tid] = acc; scnt[tid] = cnt;
    __syncthreads();
    for (int s = 128; s > 0; s >>= 1) {
        if (tid < s) { ssum[tid] += ssum[tid + s]; scnt[tid] += scnt[tid + s]; }
        __syncthreads();
    }
    if (tid == 0) {
        int last_idx = scnt[0] - 1;
        int first = ent[(size_t)b * SEQ];
        int last  = ent[(size_t)b * SEQ + last_idx];
        score[b] = Tr[BOS_ID * NUM_E + first]
                 + em[((size_t)b * SEQ) * NUM_E + first]
                 + ssum[0]
                 + Tr[last * NUM_E + EOS_ID];
    }
}

// ---------------------------------------------------------------------------
// out[0] = -mean(score - logz)
// ---------------------------------------------------------------------------
__launch_bounds__(256)
__global__ void crf_reduce(const float* __restrict__ score,
                           const float* __restrict__ logz,
                           float* __restrict__ out) {
    __shared__ float ssum[256];
    const int tid = threadIdx.x;
    float a = 0.0f;
    for (int b = tid; b < BATCH; b += 256) a += score[b] - logz[b];
    ssum[tid] = a;
    __syncthreads();
    for (int s = 128; s > 0; s >>= 1) {
        if (tid < s) ssum[tid] += ssum[tid + s];
        __syncthreads();
    }
    if (tid == 0) out[0] = -ssum[0] / (float)BATCH;
}

extern "C" void kernel_launch(void* const* d_in, const int* in_sizes, int n_in,
                              void* d_out, int out_size, void* d_ws, size_t ws_size,
                              hipStream_t stream) {
    const float*         em   = (const float*)d_in[0];
    const float*         Tr   = (const float*)d_in[1];
    const int*           ent  = (const int*)d_in[2];
    const unsigned char* mask = (const unsigned char*)d_in[3];

    float* logz  = (float*)d_ws;
    float* score = logz + BATCH;

    crf_forward_scan<<<BATCH / 16, 32, 0, stream>>>(em, Tr, logz);
    crf_gold_score<<<BATCH, 256, 0, stream>>>(em, Tr, ent, mask, score);
    crf_reduce<<<1, 256, 0, stream>>>(score, logz, (float*)d_out);
}
